// TDCPQueryAttentionPool_5729486373239
// MI455X (gfx1250) — matrix operand order repaired and checked by reading, not compiled, not run on hardware
//
#include <hip/hip_runtime.h>

typedef __bf16 bf16_t;
typedef __bf16 v16bf __attribute__((ext_vector_type(16)));
typedef __bf16 v8bf  __attribute__((ext_vector_type(8)));
typedef float  v8f   __attribute__((ext_vector_type(8)));

#define E_DIM 512
#define M_DIM 4
#define H_DIM 8
#define D_DIM 64
#define TILE  32

// workspace byte offsets
#define WS_QH    0          //  512 f32
#define WS_OBIAS 2048       //  512 f32
#define WS_CKB   4096       //  16 KB  bf16 [kt16][lane32][16]
#define WS_WVB   20480      // 512 KB  bf16 [h8][nt4][kt16][lane32][16]
#define WS_WOB   544768     // 512 KB  bf16 [h8][nt32][kt2][lane32][16]

// ---------------- prep kernels ----------------

// qh = (Wq @ q + bq) * (1/sqrt(64));  obias = Wo @ bv + bo
__global__ void prep_qh_obias(const float* __restrict__ qp,
                              const float* __restrict__ ipw,
                              const float* __restrict__ ipb,
                              const float* __restrict__ opw,
                              const float* __restrict__ opb,
                              float* __restrict__ qh,
                              float* __restrict__ obias) {
  for (int r = threadIdx.x; r < E_DIM; r += 256) {
    float acc = ipb[r];
    for (int e = 0; e < E_DIM; ++e) acc += ipw[r * E_DIM + e] * qp[e];
    qh[r] = acc * 0.125f;  // 1/sqrt(64)
    const float* bv = ipb + 2 * E_DIM;
    float ob = opb[r];
    for (int e = 0; e < E_DIM; ++e) ob += opw[r * E_DIM + e] * bv[e];
    obias[r] = ob;
  }
}

// ck[h][e] = sum_d qh[h*64+d] * Wk[h*64+d][e]; stored in WMMA-B tile order.
// B packing (contiguous k, matches load_A16): lane l holds col n=l&15; half j -> k = kt*32 + (l>>4)*16 + j
__global__ void prep_ck(const float* __restrict__ qh,
                        const float* __restrict__ ipw,
                        bf16_t* __restrict__ ckB) {
  int idx = blockIdx.x * 256 + threadIdx.x;   // 32 blocks -> 8192
  if (idx >= 16 * 32 * 16) return;
  int j = idx & 15, lane = (idx >> 4) & 31, kt = idx >> 9;
  int n = lane & 15;
  float acc = 0.f;
  if (n < H_DIM) {
    int k = kt * 32 + (lane >> 4) * 16 + j;
    const float* wk = ipw + (long)(E_DIM + n * D_DIM) * E_DIM;  // Wk rows
    for (int d = 0; d < D_DIM; ++d) acc += qh[n * D_DIM + d] * wk[(long)d * E_DIM + k];
  }
  ckB[idx] = (bf16_t)acc;
}

// Wv (in_proj rows 1024..1535) into B layout per head: [h][nt4][kt16][lane][16]
__global__ void prep_wv(const float* __restrict__ ipw, bf16_t* __restrict__ wvB) {
  int idx = blockIdx.x * 256 + threadIdx.x;   // 1024 blocks -> 262144
  int j = idx & 15, lane = (idx >> 4) & 31, kt = (idx >> 9) & 15;
  int nt = (idx >> 13) & 3, h = idx >> 15;
  int n = nt * 16 + (lane & 15);
  int k = kt * 32 + (lane >> 4) * 16 + j;
  wvB[idx] = (bf16_t)ipw[(long)(2 * E_DIM + h * D_DIM + n) * E_DIM + k];
}

// WoB[h][k=dd][n=f] = out_proj_w[f][h*64+dd] in B layout: [h][nt32][kt2][lane][16]
__global__ void prep_wo(const float* __restrict__ opw, bf16_t* __restrict__ woB) {
  int idx = blockIdx.x * 256 + threadIdx.x;   // 1024 blocks -> 262144
  int j = idx & 15, lane = (idx >> 4) & 31, kt = (idx >> 9) & 1;
  int nt = (idx >> 10) & 31, h = idx >> 15;
  int n = nt * 16 + (lane & 15);
  int kk = kt * 32 + (lane >> 4) * 16 + j;    // 0..63
  woB[idx] = (bf16_t)opw[(long)n * E_DIM + h * D_DIM + kk];
}

// ---------------- main fused kernel ----------------

__device__ __forceinline__ v16bf load_A16(const bf16_t* mat, int row_stride, int kt, int lane) {
  // A 16-bit 16x32 layout: lane row = l&15; kh = l>>4;
  // halves 0..15 -> k = kt*32 + kh*16 + 0..15 (contiguous, matches B packing)
  const int row = lane & 15, kh = lane >> 4;
  const bf16_t* p = mat + row * row_stride + kt * 32 + kh * 16;
  union { v16bf v; v8bf h[2]; } u;
  u.h[0] = *(const v8bf*)p;
  u.h[1] = *(const v8bf*)(p + 8);
  return u.v;
}

__device__ __forceinline__ v8f wmma_bf16(v16bf a, v16bf b, v8f c) {
  return __builtin_amdgcn_wmma_f32_16x16x32_bf16(false, a, false, b, (short)0, c, false, false);
}

__global__ __launch_bounds__(256, 1) void fused_attn_pool(
    const float* __restrict__ f0, const float* __restrict__ f1,
    const float* __restrict__ f2, const float* __restrict__ f3,
    const bf16_t* __restrict__ ckB, const bf16_t* __restrict__ wvB,
    const bf16_t* __restrict__ woB, const float* __restrict__ obias,
    const float* __restrict__ ln_g, const float* __restrict__ ln_b,
    float* __restrict__ out) {
  __shared__ bf16_t Xs[TILE * M_DIM * E_DIM];   // 128 KB, [t][m][e]
  __shared__ float  Ss[2 * 4 * 16 * 8];         // scores [rt][m][tr][h]
  __shared__ float  Aw[TILE * H_DIM * M_DIM];   // attn [t][h][m]
  __shared__ bf16_t Ps[2 * 16 * D_DIM];         // P tiles [rt][tr][dd]

  const int tid  = threadIdx.x;
  const int lane = tid & 31;
  const int w    = tid >> 5;
  const long tt  = (long)blockIdx.x * TILE;

  // ---- load X tile -> LDS bf16 ----
  for (int c = tid; c < TILE * M_DIM * E_DIM / 8; c += 256) {
    const int tl = c >> 8;                      // 256 vec8-chunks per token
    const int rem = c & 255;
    const int m = rem >> 6;
    const int e = (rem & 63) << 3;
    const float* src = (m == 0 ? f0 : m == 1 ? f1 : m == 2 ? f2 : f3)
                       + (tt + tl) * E_DIM + e;
    const float4 a = *(const float4*)src;
    const float4 b = *(const float4*)(src + 4);
    v8bf h;
    h[0] = (bf16_t)a.x; h[1] = (bf16_t)a.y; h[2] = (bf16_t)a.z; h[3] = (bf16_t)a.w;
    h[4] = (bf16_t)b.x; h[5] = (bf16_t)b.y; h[6] = (bf16_t)b.z; h[7] = (bf16_t)b.w;
    *(v8bf*)&Xs[(tl * M_DIM + m) * E_DIM + e] = h;
  }
  __syncthreads();

  const int rt = w >> 2;   // row-tile (16 tokens)
  const int q  = w & 3;    // quarter within row-tile

  const v8f vzero = {0.f, 0.f, 0.f, 0.f, 0.f, 0.f, 0.f, 0.f};

  // ---- scores: wave handles (rt, m=q): S = X_m[16x512] @ ck^T[512x16] ----
  {
    const int m = q;
    v8f s = vzero;
    const bf16_t* amat = &Xs[(rt * 16 * M_DIM + m) * E_DIM];
    for (int kt = 0; kt < 16; ++kt) {
      v16bf A = load_A16(amat, M_DIM * E_DIM, kt, lane);
      v16bf B = *(const v16bf*)(ckB + (kt * 32 + lane) * 16);
      s = wmma_bf16(A, B, s);
    }
    const int col = lane & 15;
    if (col < H_DIM) {
#pragma unroll
      for (int r = 0; r < 8; ++r) {
        const int trow = r + (lane >> 4) * 8;
        Ss[((rt * 4 + m) * 16 + trow) * 8 + col] = s[r];
      }
    }
  }
  __syncthreads();

  // ---- softmax over M=4 per (token, head) ----
  {
    const int t = tid >> 3, h = tid & 7;
    const int rt2 = t >> 4, tr = t & 15;
    const float s0 = Ss[((rt2 * 4 + 0) * 16 + tr) * 8 + h];
    const float s1 = Ss[((rt2 * 4 + 1) * 16 + tr) * 8 + h];
    const float s2 = Ss[((rt2 * 4 + 2) * 16 + tr) * 8 + h];
    const float s3 = Ss[((rt2 * 4 + 3) * 16 + tr) * 8 + h];
    const float mx = fmaxf(fmaxf(s0, s1), fmaxf(s2, s3));
    const float e0 = __expf(s0 - mx), e1 = __expf(s1 - mx);
    const float e2 = __expf(s2 - mx), e3 = __expf(s3 - mx);
    const float inv = 1.0f / (e0 + e1 + e2 + e3);
    float4 a4; a4.x = e0 * inv; a4.y = e1 * inv; a4.z = e2 * inv; a4.w = e3 * inv;
    *(float4*)&Aw[(t * 8 + h) * 4] = a4;
  }
  __syncthreads();

  // ---- per-head: V GEMM -> attn combine -> out-proj GEMM ----
  v8f oacc[8];
#pragma unroll
  for (int i = 0; i < 8; ++i) oacc[i] = vzero;

  for (int h = 0; h < H_DIM; ++h) {
    v8f vacc[4];
#pragma unroll
    for (int m = 0; m < 4; ++m) vacc[m] = vzero;
    const bf16_t* wvb = wvB + (long)(h * 4 + q) * (16 * 32 * 16);
    for (int kt = 0; kt < 16; ++kt) {
      v16bf B = *(const v16bf*)(wvb + (kt * 32 + lane) * 16);
#pragma unroll
      for (int m = 0; m < 4; ++m) {
        v16bf A = load_A16(&Xs[(rt * 16 * M_DIM + m) * E_DIM], M_DIM * E_DIM, kt, lane);
        vacc[m] = wmma_bf16(A, B, vacc[m]);
      }
    }
    // attn-weighted combine over m, write P (bf16) in row-major for A reload
    {
      const int kh = lane >> 4, col = lane & 15;
#pragma unroll
      for (int r = 0; r < 8; ++r) {
        const int trow = r + kh * 8;
        const int t = rt * 16 + trow;
        const float4 a4 = *(const float4*)&Aw[(t * 8 + h) * 4];
        const float p = a4.x * vacc[0][r] + a4.y * vacc[1][r] +
                        a4.z * vacc[2][r] + a4.w * vacc[3][r];
        Ps[(rt * 16 + trow) * D_DIM + q * 16 + col] = (bf16_t)p;
      }
    }
    __syncthreads();
    // out-proj: O[16x128 per wave] += P[16x64] @ WoB_h[64x512]
    const bf16_t* wob = woB + (long)h * (32 * 2 * 32 * 16);
#pragma unroll
    for (int kt = 0; kt < 2; ++kt) {
      v16bf A = load_A16(&Ps[rt * 16 * D_DIM], D_DIM, kt, lane);
#pragma unroll
      for (int i = 0; i < 8; ++i) {
        const int nt = q * 8 + i;
        v16bf B = *(const v16bf*)(wob + ((nt * 2 + kt) * 32 + lane) * 16);
        oacc[i] = wmma_bf16(A, B, oacc[i]);
      }
    }
    __syncthreads();
  }

  // ---- spill O (f32) to LDS, reusing Xs ----
  float* Os = (float*)Xs;
  {
    const int kh = lane >> 4, col0 = lane & 15;
#pragma unroll
    for (int i = 0; i < 8; ++i) {
      const int col = q * 128 + i * 16 + col0;
#pragma unroll
      for (int r = 0; r < 8; ++r) {
        const int trow = r + kh * 8;
        Os[(rt * 16 + trow) * E_DIM + col] = oacc[i][r];
      }
    }
  }
  __syncthreads();

  // ---- LayerNorm (8 threads/token) + store ----
  {
    const int t = tid >> 3, sub = tid & 7;
    const float* orow = Os + t * E_DIM + sub * 64;
    const float* obp  = obias + sub * 64;
    float s1 = 0.f, s2 = 0.f;
#pragma unroll
    for (int i = 0; i < 16; ++i) {
      float4 v = *(const float4*)(orow + i * 4);
      const float4 ob = *(const float4*)(obp + i * 4);
      v.x += ob.x; v.y += ob.y; v.z += ob.z; v.w += ob.w;
      s1 += v.x + v.y + v.z + v.w;
      s2 += v.x * v.x + v.y * v.y + v.z * v.z + v.w * v.w;
    }
#pragma unroll
    for (int off = 1; off < 8; off <<= 1) {
      s1 += __shfl_xor(s1, off, 8);
      s2 += __shfl_xor(s2, off, 8);
    }
    const float mu  = s1 * (1.0f / 512.0f);
    const float var = s2 * (1.0f / 512.0f) - mu * mu;
    const float rs  = rsqrtf(var + 1e-5f);
    float* dst = out + (tt + t) * E_DIM + sub * 64;
    const float* gp = ln_g + sub * 64;
    const float* bp = ln_b + sub * 64;
#pragma unroll
    for (int i = 0; i < 16; ++i) {
      float4 v = *(const float4*)(orow + i * 4);
      const float4 ob = *(const float4*)(obp + i * 4);
      const float4 g = *(const float4*)(gp + i * 4);
      const float4 b = *(const float4*)(bp + i * 4);
      float4 o;
      o.x = (v.x + ob.x - mu) * rs * g.x + b.x;
      o.y = (v.y + ob.y - mu) * rs * g.y + b.y;
      o.z = (v.z + ob.z - mu) * rs * g.z + b.z;
      o.w = (v.w + ob.w - mu) * rs * g.w + b.w;
      *(float4*)(dst + i * 4) = o;
    }
  }
}

// ---------------- launch ----------------

extern "C" void kernel_launch(void* const* d_in, const int* in_sizes, int n_in,
                              void* d_out, int out_size, void* d_ws, size_t ws_size,
                              hipStream_t stream) {
  (void)in_sizes; (void)n_in; (void)out_size; (void)ws_size;
  const float* f0  = (const float*)d_in[0];
  const float* f1  = (const float*)d_in[1];
  const float* f2  = (const float*)d_in[2];
  const float* f3  = (const float*)d_in[3];
  const float* qp  = (const float*)d_in[4];
  const float* ipw = (const float*)d_in[5];
  const float* ipb = (const float*)d_in[6];
  const float* opw = (const float*)d_in[7];
  const float* opb = (const float*)d_in[8];
  const float* lng = (const float*)d_in[9];
  const float* lnb = (const float*)d_in[10];

  char* ws = (char*)d_ws;
  float*  qh    = (float*)(ws + WS_QH);
  float*  obias = (float*)(ws + WS_OBIAS);
  bf16_t* ckB   = (bf16_t*)(ws + WS_CKB);
  bf16_t* wvB   = (bf16_t*)(ws + WS_WVB);
  bf16_t* woB   = (bf16_t*)(ws + WS_WOB);

  prep_qh_obias<<<1, 256, 0, stream>>>(qp, ipw, ipb, opw, opb, qh, obias);
  prep_ck<<<32, 256, 0, stream>>>(qh, ipw, ckB);
  prep_wv<<<1024, 256, 0, stream>>>(ipw, wvB);
  prep_wo<<<1024, 256, 0, stream>>>(opw, woB);

  const int n_tiles = (8 * 4096) / TILE;  // 1024
  fused_attn_pool<<<n_tiles, 256, 0, stream>>>(f0, f1, f2, f3, ckB, wvB, woB,
                                               obias, lng, lnb, (float*)d_out);
}
